// SelectiveScan_48765058678852
// MI455X (gfx1250) — matrix operand order repaired and checked by reading, not compiled, not run on hardware
//
#include <hip/hip_runtime.h>

// ---------------------------------------------------------------------------
// SelectiveScan for MI455X (gfx1250, wave32).
//   B=8, L=2048, D=1024, N=16
// Pipeline:
//   1) cvt_f32_bf16   : x, Wg -> bf16 copies in workspace (RNE)
//   2) gate_gemm      : TDM tensor_load_to_lds double-buffered tile feed +
//                       v_wmma_f32_16x16x32_bf16, fused sigmoid*x, epilogue
//                       transposed through LDS and written by one
//                       TENSOR_STORE_FROM_LDS per block -> xgT[d][b][l]
//   3) ssm_scan       : 1 WG per channel d, 4 waves x 2 batches,
//                       shfl-broadcast 16x16 recurrence, writes yresT[d][b][l]
//   4) ln_out         : transposing LayerNorm -> out[b][l][d]
// Workspace: 128 MB  (xb 32M | wgb 2M | pad | xgT 64M ; yresT reuses [0,64M))
// ---------------------------------------------------------------------------

#define DI 1024
#define NI 16
#define BI 8
#define LI 2048
#define MTOT (BI * LI)   // 16384

typedef __bf16 bf16_t;
typedef __attribute__((ext_vector_type(16))) __bf16 v16bf;
typedef __attribute__((ext_vector_type(8)))  float  v8f;

union Frag {               // 32B WMMA operand: two 16B LDS loads -> v16bf
  uint4 q[2];
  v16bf f;
};

__device__ __forceinline__ unsigned short f2bf(float f) {
  unsigned int u = __float_as_uint(f);
  u += 0x7fffu + ((u >> 16) & 1u);           // round-to-nearest-even
  return (unsigned short)(u >> 16);
}

// ------------------------- TDM (Tensor Data Mover) -------------------------
typedef __attribute__((ext_vector_type(4))) unsigned int tdm_u4;
typedef __attribute__((ext_vector_type(4))) int          tdm_i4;
typedef __attribute__((ext_vector_type(8))) int          tdm_i8;

#if __has_builtin(__builtin_amdgcn_tensor_load_to_lds) && \
    __has_builtin(__builtin_amdgcn_tensor_store_from_lds)
#define HAVE_TDM 1

__device__ __forceinline__ void tdm_load(tdm_u4 g0, tdm_i8 g1) {
  tdm_i4 z4 = {0, 0, 0, 0};
#if __clang_major__ >= 23
  tdm_i8 z8 = {0, 0, 0, 0, 0, 0, 0, 0};
  __builtin_amdgcn_tensor_load_to_lds(g0, g1, z4, z4, z8, 0);
#else
  __builtin_amdgcn_tensor_load_to_lds(g0, g1, z4, z4, 0);
#endif
}

__device__ __forceinline__ void tdm_store(tdm_u4 g0, tdm_i8 g1) {
  tdm_i4 z4 = {0, 0, 0, 0};
#if __clang_major__ >= 23
  tdm_i8 z8 = {0, 0, 0, 0, 0, 0, 0, 0};
  __builtin_amdgcn_tensor_store_from_lds(g0, g1, z4, z4, z8, 0);
#else
  __builtin_amdgcn_tensor_store_from_lds(g0, g1, z4, z4, 0);
#endif
}

// D# group 0: count=1 | lds_addr | global byte address (57b) | type=2
__device__ __forceinline__ tdm_u4 tdm_g0(unsigned lds_addr,
                                         unsigned long long gaddr) {
  tdm_u4 g;
  g.x = 1u;                                      // count=1, user mode
  g.y = lds_addr;                                // LDS byte address
  g.z = (unsigned)gaddr;                         // addr[31:0]
  g.w = ((unsigned)(gaddr >> 32) & 0x01FFFFFFu)  // addr[56:32]
        | (2u << 30);                            // type=2 ("image")
  return g;
}

// D# group 1 for a 2-D bf16 tile; pads each 16-DWORD LDS row with 4 DWORDs
// (32 bf16 tile row -> 40 bf16 LDS row, matches LDA/LDB bank-conflict pad).
__device__ __forceinline__ tdm_i8 tdm_g1_bf16(int dim0, int dim1, int tile0,
                                              int tile1, int stride0) {
  tdm_i8 g;
  g[0] = (1 << 16)                 // data_size = 2 bytes
         | (1 << 20)               // pad_enable (load-only)
         | (3 << 22)               // pad_interval: 16 DWORDs
         | (3 << 25);              // pad_amount:    4 DWORDs
  g[1] = (dim0 & 0xFFFF) << 16;                      // tensor_dim0[15:0]
  g[2] = (int)(((unsigned)dim0 >> 16) | ((unsigned)(dim1 & 0xFFFF) << 16));
  g[3] = (int)(((unsigned)dim1 >> 16) | ((unsigned)tile0 << 16));
  g[4] = tile1;                    // tile_dim1 | tile_dim2=0
  g[5] = stride0;                  // tensor_dim0_stride[31:0]
  g[6] = 0;
  g[7] = 0;
  return g;
}

// D# group 1 for a 2-D fp32 tile, no padding (store path).
__device__ __forceinline__ tdm_i8 tdm_g1_f32(int dim0, int dim1, int tile0,
                                             int tile1, int stride0) {
  tdm_i8 g;
  g[0] = (2 << 16);                // data_size = 4 bytes
  g[1] = (dim0 & 0xFFFF) << 16;
  g[2] = (int)(((unsigned)dim0 >> 16) | ((unsigned)(dim1 & 0xFFFF) << 16));
  g[3] = (int)(((unsigned)dim1 >> 16) | ((unsigned)tile0 << 16));
  g[4] = tile1;
  g[5] = stride0;
  g[6] = 0;
  g[7] = 0;
  return g;
}
#endif  // HAVE_TDM

// -------------------------------- 1) convert -------------------------------
__global__ __launch_bounds__(256) void cvt_f32_bf16(
    const float* __restrict__ src, unsigned short* __restrict__ dst, int n) {
  int i = (blockIdx.x * 256 + threadIdx.x) * 4;
  if (i + 3 < n) {
    float4 v = *(const float4*)(src + i);
    ushort4 o;
    o.x = f2bf(v.x); o.y = f2bf(v.y); o.z = f2bf(v.z); o.w = f2bf(v.w);
    *(ushort4*)(dst + i) = o;
  }
}

// -------------------------------- 2) gate GEMM -----------------------------
// C[m,e] = sum_d x[m,d] * Wg[e,d];  xg = x * sigmoid(C + bg)
// Block tile 128(M) x 64(N), K-step 32, 8 waves in a 4x2 grid, each wave
// owns a 32x32 patch = 2x2 wmma 16x16 tiles. LDS tiles are double-buffered
// and fed by TDM (wave 0 issues, all waves compute, one barrier per K-step).
// Epilogue transposes through LDS and DMA-stores the 64(d) x 128(l) tile.
#define BM 128
#define BN 64
#define BK 32
#define LDA 40   // padded LDS row (bf16 units): 16 rows -> distinct banks
#define LDB 40

// Fallback staging when the TDM builtin is unavailable (also host pass).
__device__ __forceinline__ void stage_tiles(
    const unsigned short* __restrict__ xb, const unsigned short* __restrict__ wgb,
    unsigned short* As, unsigned short* Bs, int m0, int n0, int k0, int tid) {
#pragma unroll
  for (int p = 0; p < 2; ++p) {
    int idx = tid + p * 256;                 // 0..511
    int row = idx >> 2, seg = idx & 3;
    uint4 v = *(const uint4*)(xb + (size_t)(m0 + row) * DI + k0 + seg * 8);
    *(uint4*)(As + row * LDA + seg * 8) = v;
  }
  int row = tid >> 2, seg = tid & 3;
  uint4 v = *(const uint4*)(wgb + (size_t)(n0 + row) * DI + k0 + seg * 8);
  *(uint4*)(Bs + row * LDB + seg * 8) = v;
}

__global__ __launch_bounds__(256) void gate_gemm(
    const unsigned short* __restrict__ xb,    // bf16 [MTOT][DI]
    const unsigned short* __restrict__ wgb,   // bf16 [DI][DI]  (e, d)
    const float* __restrict__ x,              // fp32 [MTOT][DI]
    const float* __restrict__ bg,             // [DI]
    float* __restrict__ xgT) {                // [DI][BI][LI]
  __shared__ __align__(16) unsigned short As[2][BM * LDA];  // 2 x 10 KB
  __shared__ __align__(16) unsigned short Bs[2][BN * LDB];  // 2 x  5 KB
  __shared__ __align__(16) float Cs[BN * BM];               // 32 KB out tile

  const int m0 = blockIdx.x * BM;
  const int n0 = blockIdx.y * BN;
  const int tid = threadIdx.x;
  const int wid = tid >> 5, lane = tid & 31;
  const int wm = wid >> 1, wn = wid & 1;     // 4x2 wave grid
  const int lhalf = lane >> 4, lmod = lane & 15;

  v8f acc[2][2];
#pragma unroll
  for (int i = 0; i < 2; ++i)
#pragma unroll
    for (int j = 0; j < 2; ++j) {
      v8f z = {0.f, 0.f, 0.f, 0.f, 0.f, 0.f, 0.f, 0.f};
      acc[i][j] = z;
    }

#if HAVE_TDM
  // Uniform descriptor state (kept in SGPRs; only global_addr moves with k).
  const unsigned ldsA[2] = {(unsigned)(unsigned long long)(uintptr_t)&As[0][0],
                            (unsigned)(unsigned long long)(uintptr_t)&As[1][0]};
  const unsigned ldsB[2] = {(unsigned)(unsigned long long)(uintptr_t)&Bs[0][0],
                            (unsigned)(unsigned long long)(uintptr_t)&Bs[1][0]};
  const unsigned long long gA0 =
      (unsigned long long)(uintptr_t)xb + (unsigned long long)m0 * DI * 2ull;
  const unsigned long long gB0 =
      (unsigned long long)(uintptr_t)wgb + (unsigned long long)n0 * DI * 2ull;
  const tdm_i8 dA = tdm_g1_bf16(DI, MTOT, BK, BM, DI);  // 128x32 bf16 tile
  const tdm_i8 dB = tdm_g1_bf16(DI, DI,   BK, BN, DI);  //  64x32 bf16 tile

  if (wid == 0) {                       // prologue: DMA buffer 0 (k0 = 0)
    tdm_load(tdm_g0(ldsA[0], gA0), dA);
    tdm_load(tdm_g0(ldsB[0], gB0), dB);
    __builtin_amdgcn_s_wait_tensorcnt(0);
  }
  __syncthreads();
#else
  stage_tiles(xb, wgb, &As[0][0], &Bs[0][0], m0, n0, 0, tid);
  __syncthreads();
#endif

  for (int ks = 0; ks < DI / BK; ++ks) {
    const int cur = ks & 1;
    const int k0 = ks * BK;

    // kick off next tiles while this buffer is consumed by WMMAs
#if HAVE_TDM
    if ((ks + 1) < DI / BK && wid == 0) {
      unsigned long long koff = (unsigned long long)(k0 + BK) * 2ull;
      tdm_load(tdm_g0(ldsA[cur ^ 1], gA0 + koff), dA);
      tdm_load(tdm_g0(ldsB[cur ^ 1], gB0 + koff), dB);
    }
#else
    if ((ks + 1) < DI / BK)
      stage_tiles(xb, wgb, &As[cur ^ 1][0], &Bs[cur ^ 1][0], m0, n0, k0 + BK,
                  tid);
#endif
    if (k0 + 2 * BK < DI) {  // global_prefetch_b8 two K-steps ahead
      __builtin_prefetch(xb + (size_t)(m0 + (tid >> 1)) * DI + k0 + 2 * BK +
                             (tid & 1) * 16, 0, 1);
      __builtin_prefetch(wgb + (size_t)(n0 + (tid >> 2)) * DI + k0 + 2 * BK,
                         0, 1);
    }

#pragma unroll
    for (int mt = 0; mt < 2; ++mt) {
      // A frag: lane -> M = lmod, K = lhalf*8 + {0..7} in v0..3, +16 in v4..7
      const unsigned short* ap =
          &As[cur][(wm * 32 + mt * 16 + lmod) * LDA + lhalf * 8];
      Frag fa;
      fa.q[0] = *(const uint4*)ap;
      fa.q[1] = *(const uint4*)(ap + 16);
#pragma unroll
      for (int nt = 0; nt < 2; ++nt) {
        // B frag: lane -> N = lmod, K = lhalf*8 + {0..7} in v0..3, +16 in v4..7
        const unsigned short* bptr =
            &Bs[cur][(wn * 32 + nt * 16 + lmod) * LDB + lhalf * 8];
        Frag fb;
        fb.q[0] = *(const uint4*)bptr;
        fb.q[1] = *(const uint4*)(bptr + 16);
        acc[mt][nt] = __builtin_amdgcn_wmma_f32_16x16x32_bf16(
            false, fa.f, false, fb.f, (short)0, acc[mt][nt], false, false);
      }
    }

#if HAVE_TDM
    if (wid == 0) __builtin_amdgcn_s_wait_tensorcnt(0);  // next buffer landed
#endif
    __syncthreads();
  }

  // Epilogue: sigmoid gate * fp32 x, transpose into Cs[n_local][m_local],
  // then one TDM store of the 64(d) x 128(l) tile into xgT.
#pragma unroll
  for (int mt = 0; mt < 2; ++mt)
#pragma unroll
    for (int nt = 0; nt < 2; ++nt) {
      int nl = wn * 32 + nt * 16 + lmod;
      int n = n0 + nl;
      float bgv = bg[n];
#pragma unroll
      for (int j = 0; j < 8; ++j) {
        int ml = wm * 32 + mt * 16 + j + 8 * lhalf;   // C/D layout
        int m = m0 + ml;
        float g = acc[mt][nt][j] + bgv;
        float gate = 1.f / (1.f + __expf(-g));
        float xv = x[(size_t)m * DI + n];
        Cs[nl * BM + ml] = xv * gate;
      }
    }
  __syncthreads();

  const int bidx = m0 >> 11;            // batch of this 128-row M band
  const int l0 = m0 & (LI - 1);         // l offset inside the batch
#if HAVE_TDM
  if (wid == 0) {
    unsigned long long gC =
        (unsigned long long)(uintptr_t)xgT +
        ((unsigned long long)n0 * MTOT + (unsigned long long)bidx * LI + l0) *
            4ull;
    tdm_store(tdm_g0((unsigned)(unsigned long long)(uintptr_t)&Cs[0], gC),
              tdm_g1_f32(MTOT, DI, BM, BN, MTOT));
    __builtin_amdgcn_s_wait_tensorcnt(0);
  }
#else
  for (int idx = tid; idx < BN * BM / 4; idx += 256) {
    int nl = idx / (BM / 4);
    int ms = (idx % (BM / 4)) * 4;
    *(float4*)(xgT + (size_t)(n0 + nl) * MTOT + (size_t)bidx * LI + l0 + ms) =
        *(const float4*)(Cs + nl * BM + ms);
  }
#endif
}

// -------------------------------- 3) scan ----------------------------------
// One workgroup per channel d (1024 WGs, 128 threads = 4 waves).
// Wave w owns batches {2w, 2w+1}: lane = 16*g + m, g = batch-half, m = state.
// Recurrence: s[m] <- sum_n shfl(s, n)*A[n][m] + xt*Bp[m]
//             y = reduce_m(s[m]*Cp[m]) + (Dp+1)*xt   (residual fused)
__global__ __launch_bounds__(128) void ssm_scan(
    const float* __restrict__ A,    // [DI][NI][NI]
    const float* __restrict__ Bp,   // [DI][NI]
    const float* __restrict__ Cp,   // [DI][NI]
    const float* __restrict__ Dp,   // [DI]
    const float* __restrict__ xgT,  // [DI][BI][LI]
    float* __restrict__ yT) {       // [DI][BI][LI]
  const int d = blockIdx.x;
  const int tid = threadIdx.x;
  const int w = tid >> 5, lane = tid & 31;
  const int g = lane >> 4, m = lane & 15;

  __shared__ float xbuf[4][2][32];
  __shared__ float ybuf[4][2][32];

  float a[NI];  // column m of A[d] (i.e. A[d][n][m] for n = 0..15)
#pragma unroll
  for (int n = 0; n < NI; ++n) a[n] = A[(size_t)d * 256 + n * 16 + m];
  const float bp = Bp[d * 16 + m];
  const float cp = Cp[d * 16 + m];
  const float dpr = Dp[d] + 1.f;   // fold in the +xg residual

  const float* xrow0 = xgT + ((size_t)d * BI + 2 * w) * LI;
  const float* xrow1 = xgT + ((size_t)d * BI + 2 * w + 1) * LI;
  float* yrow0 = yT + ((size_t)d * BI + 2 * w) * LI;
  float* yrow1 = yT + ((size_t)d * BI + 2 * w + 1) * LI;

  float s = 0.f;
  for (int t0 = 0; t0 < LI; t0 += 32) {
    xbuf[w][0][lane] = xrow0[t0 + lane];   // 128B coalesced per batch
    xbuf[w][1][lane] = xrow1[t0 + lane];
    for (int tt = 0; tt < 32; ++tt) {
      float xt = xbuf[w][g][tt];
      float acc0 = xt * bp;                // two accumulators for ILP
      float acc1 = 0.f;
#pragma unroll
      for (int n = 0; n < 8; ++n) {
        acc0 += __shfl(s, n, 16) * a[n];
        acc1 += __shfl(s, n + 8, 16) * a[n + 8];
      }
      s = acc0 + acc1;
      float p = s * cp;                    // butterfly-reduce over 16 lanes
      p += __shfl_xor(p, 8, 16);
      p += __shfl_xor(p, 4, 16);
      p += __shfl_xor(p, 2, 16);
      p += __shfl_xor(p, 1, 16);
      if (m == 0) ybuf[w][g][tt] = p + dpr * xt;
    }
    yrow0[t0 + lane] = ybuf[w][0][lane];   // 128B coalesced flush
    yrow1[t0 + lane] = ybuf[w][1][lane];
  }
}

// -------------------------------- 4) LayerNorm -----------------------------
// Block = (b, 32 l's). Pass 1: stats streaming yT (channel-major, coalesced
// 128B per d). Pass 2: re-read (L2-resident), normalize, write out[b][l][d]
// as float4 (128B per l across the 8 d-groups).
__global__ __launch_bounds__(256) void ln_out(
    const float* __restrict__ yT,     // [DI][BI][LI]
    const float* __restrict__ gamma,  // [DI]
    const float* __restrict__ beta,   // [DI]
    float* __restrict__ out) {        // [BI][LI][DI]
  const int b = blockIdx.y;
  const int l0 = blockIdx.x * 32;
  const int tid = threadIdx.x;
  const int l = tid & 31, dg = tid >> 5;  // dg in 0..7

  __shared__ float ssum[8][32], ssq[8][32], smu[32], srs[32];

  float sum = 0.f, sq = 0.f;
  for (int d = dg; d < DI; d += 8) {
    float v = yT[((size_t)d * BI + b) * LI + l0 + l];
    sum += v;
    sq += v * v;
  }
  ssum[dg][l] = sum;
  ssq[dg][l] = sq;
  __syncthreads();
  if (dg == 0) {
    float ts = 0.f, tq = 0.f;
#pragma unroll
    for (int i = 0; i < 8; ++i) { ts += ssum[i][l]; tq += ssq[i][l]; }
    float mu = ts * (1.f / DI);
    float var = tq * (1.f / DI) - mu * mu;
    smu[l] = mu;
    srs[l] = rsqrtf(var + 1e-5f);
  }
  __syncthreads();
  const float mu = smu[l], rs = srs[l];

  for (int d4 = dg * 4; d4 < DI; d4 += 32) {
    float v0 = yT[((size_t)(d4 + 0) * BI + b) * LI + l0 + l];
    float v1 = yT[((size_t)(d4 + 1) * BI + b) * LI + l0 + l];
    float v2 = yT[((size_t)(d4 + 2) * BI + b) * LI + l0 + l];
    float v3 = yT[((size_t)(d4 + 3) * BI + b) * LI + l0 + l];
    float4 o;
    o.x = (v0 - mu) * rs * gamma[d4 + 0] + beta[d4 + 0];
    o.y = (v1 - mu) * rs * gamma[d4 + 1] + beta[d4 + 1];
    o.z = (v2 - mu) * rs * gamma[d4 + 2] + beta[d4 + 2];
    o.w = (v3 - mu) * rs * gamma[d4 + 3] + beta[d4 + 3];
    *(float4*)(out + ((size_t)b * LI + l0 + l) * DI + d4) = o;
  }
}

// -------------------------------- launcher ---------------------------------
extern "C" void kernel_launch(void* const* d_in, const int* in_sizes, int n_in,
                              void* d_out, int out_size, void* d_ws,
                              size_t ws_size, hipStream_t stream) {
  const float* x     = (const float*)d_in[0];
  const float* A     = (const float*)d_in[1];
  const float* Bp    = (const float*)d_in[2];
  const float* Cp    = (const float*)d_in[3];
  const float* Dp    = (const float*)d_in[4];
  const float* Wg    = (const float*)d_in[5];
  const float* bg    = (const float*)d_in[6];
  const float* gamma = (const float*)d_in[7];
  const float* beta  = (const float*)d_in[8];

  char* ws = (char*)d_ws;                       // needs 128 MB
  unsigned short* xb  = (unsigned short*)(ws);                  // 32 MB
  unsigned short* wgb = (unsigned short*)(ws + (32ull << 20));  //  2 MB
  float* xgT = (float*)(ws + (64ull << 20));                    // 64 MB
  float* yT  = (float*)(ws);  // reuses [0,64M): xb/wgb dead after GEMM

  cvt_f32_bf16<<<(MTOT * DI) / 1024, 256, 0, stream>>>(x, xb, MTOT * DI);
  cvt_f32_bf16<<<(DI * DI) / 1024, 256, 0, stream>>>(Wg, wgb, DI * DI);

  dim3 gg(MTOT / BM, DI / BN);  // 128 x 16
  gate_gemm<<<gg, 256, 0, stream>>>(xb, wgb, x, bg, xgT);

  ssm_scan<<<DI, 128, 0, stream>>>(A, Bp, Cp, Dp, xgT, yT);

  dim3 gl(LI / 32, BI);         // 64 x 8
  ln_out<<<gl, 256, 0, stream>>>(yT, gamma, beta, (float*)d_out);
}
